// multi_head_attention_88459146428934
// MI455X (gfx1250) — matrix operand order repaired and checked by reading, not compiled, not run on hardware
//
#include <hip/hip_runtime.h>
#include <hip/hip_bf16.h>
#include <stdint.h>

// ---------- types ----------
typedef __attribute__((ext_vector_type(16))) __bf16 v16bf;
typedef __attribute__((ext_vector_type(8)))  float  v8f;
typedef __attribute__((ext_vector_type(8)))  __bf16 bf16x8;
typedef __attribute__((ext_vector_type(4)))  __bf16 bf16x4;
typedef __attribute__((ext_vector_type(4)))  float  f32x4;
typedef __attribute__((ext_vector_type(4)))  unsigned int u32x4;
typedef __attribute__((ext_vector_type(8)))  int   i32x8;
typedef __attribute__((ext_vector_type(4)))  int   i32x4;

#define DEV static __device__ __forceinline__
#define AS3(p) ((__attribute__((address_space(3))) void*)(p))

// ---------- feature detection ----------
#if defined(__has_builtin)
#  if __has_builtin(__builtin_amdgcn_tensor_load_to_lds) && \
      __has_builtin(__builtin_amdgcn_s_wait_tensorcnt)
#    define USE_TDM 1
#  endif
#endif

// ---------- WMMA helpers ----------
DEV v8f wmma_bf16(v16bf a, v16bf b, v8f c) {
  return __builtin_amdgcn_wmma_f32_16x16x32_bf16(
      false, a, false, b, (short)0, c, false, false);
}

DEV v8f vzero8() { v8f v = {0.f,0.f,0.f,0.f,0.f,0.f,0.f,0.f}; return v; }

// A-matrix 16x32 bf16 fragment: lanes 0-15 hold K=0..7 & 16..23 of row M=lane,
// lanes 16-31 hold K=8..15 & 24..31.  `base` row-major [rows][ld] bf16.
DEV v16bf load_frag_a(const __bf16* base, int ld, int rowBase, int kOff) {
  int lane = threadIdx.x & 31;
  int r    = lane & 15;
  int kb   = (lane >> 4) << 3;
  const __bf16* p = base + (size_t)(rowBase + r) * ld + kOff + kb;
  bf16x8 lo = *(const bf16x8*)(p);
  bf16x8 hi = *(const bf16x8*)(p + 16);
  v16bf f;
#pragma unroll
  for (int i = 0; i < 8; ++i) { f[i] = lo[i]; f[i + 8] = hi[i]; }
  return f;
}

// B-matrix 32x16 bf16 fragment: lanes 0-15 hold K=0..7 & 16..23 of col N=lane,
// lanes 16-31 hold K=8..15 & 24..31.  `base` stored [col][ld] contiguous in K.
DEV v16bf load_frag_b(const __bf16* base, int ld, int colBase, int kOff) {
  int lane = threadIdx.x & 31;
  const __bf16* p = base + (size_t)(colBase + (lane & 15)) * ld + kOff
                         + ((lane >> 4) << 3);
  bf16x8 lo = *(const bf16x8*)(p);
  bf16x8 hi = *(const bf16x8*)(p + 16);
  v16bf f;
#pragma unroll
  for (int i = 0; i < 8; ++i) { f[i] = lo[i]; f[i + 8] = hi[i]; }
  return f;
}

// ---------- tile staging: TDM (fallback: sync copy) ----------
#ifdef USE_TDM
// 2D tile global->LDS via Tensor Data Mover.  tensor == tile footprint,
// dim0 stride = global row pitch, LDS padding via pad_interval/pad_amount.
// padI: pad after 2^(padI+1) DWORDs; padA: pad (padA+1) DWORDs.
DEV void tdm_load_2d(__bf16* ldsDst, const __bf16* gSrc,
                     unsigned tileW, unsigned tileH, unsigned strideElems,
                     unsigned padI, unsigned padA) {
  unsigned lds = (unsigned)(uintptr_t)AS3(ldsDst);
  unsigned long long ga = (unsigned long long)(uintptr_t)gSrc;
  u32x4 g0;
  g0[0] = 1u;                                              // count=1 user desc
  g0[1] = lds;                                             // lds_addr
  g0[2] = (unsigned)ga;                                    // global_addr lo
  g0[3] = (unsigned)((ga >> 32) & 0x1FFFFFFu) | 0x80000000u; // addr hi | type=2
  i32x8 g1;
  g1[0] = (int)((1u << 16) |                               // data_size = 2B
                (1u << 20) |                               // pad_enable
                (padI << 22) | (padA << 25));
  g1[1] = (int)((tileW & 0xFFFFu) << 16);                  // tensor_dim0 lo
  g1[2] = (int)(((tileW >> 16) & 0xFFFFu) | ((tileH & 0xFFFFu) << 16));
  g1[3] = (int)(((tileH >> 16) & 0xFFFFu) | ((tileW & 0xFFFFu) << 16)); // |tile_dim0
  g1[4] = (int)(tileH & 0xFFFFu);                          // tile_dim1
  g1[5] = (int)strideElems;                                // dim0_stride lo
  g1[6] = 0;
  g1[7] = 0;
  i32x4 z4; z4[0] = z4[1] = z4[2] = z4[3] = 0;
  i32x8 z8;
#pragma unroll
  for (int i = 0; i < 8; ++i) z8[i] = 0;
  __builtin_amdgcn_tensor_load_to_lds(g0, g1, z4, z4, z8, 0);
}
#endif

DEV void stage_wait() {
#ifdef USE_TDM
  __builtin_amdgcn_s_wait_tensorcnt(0);
#endif
}

// ---------- fp32 -> bf16 convert ----------
__global__ __launch_bounds__(256) void cvt_f32_bf16(
    const float* __restrict__ s, __bf16* __restrict__ d, int n4) {
  int i = blockIdx.x * 256 + threadIdx.x;
  if (i < n4) {
    f32x4 v = *(const f32x4*)(s + (size_t)i * 4);
    bf16x4 o;
#pragma unroll
    for (int j = 0; j < 4; ++j) o[j] = (__bf16)v[j];
    *(bf16x4*)(d + (size_t)i * 4) = o;
  }
}

// ---------- bf16 WMMA GEMM, double-buffered ----------
#define BM 128
#define BN 128
#define BK 32
#define LDT 40   // 32 data + 8 pad bf16 (80 B rows; TDM padI=3, padA=3)

__global__ __launch_bounds__(256) void gemm_bf16(
    const __bf16* __restrict__ A, const __bf16* __restrict__ Bw,
    __bf16* __restrict__ Cb, float* __restrict__ Cf, float cScale,
    int M, int N, int K) {
  __shared__ __bf16 sA[2][BM * LDT];   // [row][k]
  __shared__ __bf16 sB[2][BN * LDT];   // transposed: [col][k]
  int tid  = threadIdx.x;
  int wave = tid >> 5;
  int lane = tid & 31;
  int mBase = blockIdx.y * BM;
  int nBase = blockIdx.x * BN;
  int wm = wave & 3;
  int wn = wave >> 2;

  // B staging coords: rows br0, br0+16; 8-elem column chunk bc0
  int br0 = tid >> 4;
  int bc0 = (tid & 15) << 3;
  // A fallback staging coords: rows ar0, ar0+64; 8-elem k chunk ac0
  int ar0 = tid >> 2;
  int ac0 = (tid & 3) << 3;
  (void)ar0; (void)ac0;

  v8f acc[2][4];
#pragma unroll
  for (int i = 0; i < 2; ++i)
#pragma unroll
    for (int j = 0; j < 4; ++j) acc[i][j] = vzero8();

  bf16x8 breg0, breg1;
  auto loadB = [&](int kt) {
    breg0 = *(const bf16x8*)(Bw + (size_t)(kt + br0) * N + nBase + bc0);
    breg1 = *(const bf16x8*)(Bw + (size_t)(kt + br0 + 16) * N + nBase + bc0);
  };
  auto storeB = [&](int buf) {
#pragma unroll
    for (int j = 0; j < 8; ++j) {
      sB[buf][(bc0 + j) * LDT + br0]      = breg0[j];
      sB[buf][(bc0 + j) * LDT + br0 + 16] = breg1[j];
    }
  };
  auto stageA = [&](int buf, int kt) {
#ifdef USE_TDM
    if (wave == 0)
      tdm_load_2d(sA[buf], A + (size_t)mBase * K + kt, BK, BM, (unsigned)K, 3, 3);
#else
    const __bf16* g0p = A + (size_t)(mBase + ar0) * K + kt + ac0;
    const __bf16* g1p = A + (size_t)(mBase + ar0 + 64) * K + kt + ac0;
    *(bf16x8*)(&sA[buf][ar0 * LDT + ac0])        = *(const bf16x8*)g0p;
    *(bf16x8*)(&sA[buf][(ar0 + 64) * LDT + ac0]) = *(const bf16x8*)g1p;
#endif
  };

  // prologue: stage kt=0 into buffer 0
  loadB(0);
  stageA(0, 0);
  storeB(0);
  stage_wait();
  __syncthreads();

  int cur = 0;
  for (int kt = 0; kt < K; kt += BK) {
    bool nxt = (kt + BK) < K;
    if (nxt) {                       // launch next tile while we compute
      stageA(cur ^ 1, kt + BK);
      loadB(kt + BK);
    }
    v16bf af[2], bfm[4];
#pragma unroll
    for (int i = 0; i < 2; ++i) af[i]  = load_frag_a(sA[cur], LDT, wm * 32 + i * 16, 0);
#pragma unroll
    for (int j = 0; j < 4; ++j) bfm[j] = load_frag_b(sB[cur], LDT, wn * 64 + j * 16, 0);
#pragma unroll
    for (int i = 0; i < 2; ++i)
#pragma unroll
      for (int j = 0; j < 4; ++j)
        acc[i][j] = wmma_bf16(af[i], bfm[j], acc[i][j]);
    if (nxt) {
      storeB(cur ^ 1);
      stage_wait();
      __syncthreads();
      cur ^= 1;
    }
  }

  int col  = lane & 15;
  int half = lane >> 4;
#pragma unroll
  for (int i = 0; i < 2; ++i)
#pragma unroll
    for (int j = 0; j < 4; ++j) {
      int cg = nBase + wn * 64 + j * 16 + col;
#pragma unroll
      for (int r = 0; r < 8; ++r) {
        int rg = mBase + wm * 32 + i * 16 + r + half * 8;
        float v = acc[i][j][r] * cScale;
        if (Cf) Cf[(size_t)rg * N + cg] = v;
        else    Cb[(size_t)rg * N + cg] = (__bf16)v;
      }
    }
}

// ---------- flash attention, double-buffered K/V ----------
#define AS  2048
#define AD  1024
#define ADH 64
#define LDK 72   // 64 data + 8 pad (TDM padI=4, padA=3)
#define LDV 40
#define LDP 40

__global__ __launch_bounds__(128) void attn_fused(
    const __bf16* __restrict__ Q, const __bf16* __restrict__ Km,
    const __bf16* __restrict__ V, __bf16* __restrict__ Ctx) {
  __shared__ __bf16 sK [2][32 * LDK];     // [kpos][d]
  __shared__ __bf16 sVt[2][64 * LDV];     // [d][kpos]
  __shared__ __bf16 sP [4][32 * LDP];     // per-wave probs [q][kpos]

  int tid  = threadIdx.x;
  int wave = tid >> 5;
  int lane = tid & 31;
  int qBase = blockIdx.x * 128;
  int bh = blockIdx.y;
  int b = bh >> 4, h = bh & 15;
  size_t rowB = (size_t)b * AS;
  int hOff = h * ADH;
  int qWave = qBase + wave * 32;

  // staging coords: rows rk, rk+16; 8-elem d chunk dk
  int rk = tid >> 3;
  int dk = (tid & 7) << 3;

  // Q fragments from global (Q was pre-scaled by 1/sqrt(D) in its GEMM)
  v16bf qf[2][2];
#pragma unroll
  for (int m = 0; m < 2; ++m)
#pragma unroll
    for (int kc = 0; kc < 2; ++kc)
      qf[m][kc] = load_frag_a(Q + (rowB + qWave) * AD + hOff, AD, m * 16, kc * 32);

  v8f acc[2][4];
#pragma unroll
  for (int m = 0; m < 2; ++m)
#pragma unroll
    for (int n = 0; n < 4; ++n) acc[m][n] = vzero8();
  float rowM[2][8], rowL[2][8];
#pragma unroll
  for (int m = 0; m < 2; ++m)
#pragma unroll
    for (int r = 0; r < 8; ++r) { rowM[m][r] = -1e30f; rowL[m][r] = 0.f; }

  bf16x8 vreg0, vreg1;
  auto loadV = [&](int kt) {
    vreg0 = *(const bf16x8*)(V + (rowB + kt + rk) * AD + hOff + dk);
    vreg1 = *(const bf16x8*)(V + (rowB + kt + rk + 16) * AD + hOff + dk);
  };
  auto storeV = [&](int buf) {
#pragma unroll
    for (int j = 0; j < 8; ++j) {
      sVt[buf][(dk + j) * LDV + rk]      = vreg0[j];
      sVt[buf][(dk + j) * LDV + rk + 16] = vreg1[j];
    }
  };
  auto stageK = [&](int buf, int kt) {
#ifdef USE_TDM
    if (wave == 0)
      tdm_load_2d(sK[buf], Km + (rowB + kt) * AD + hOff, ADH, 32, AD, 4, 3);
#else
    const __bf16* g0p = Km + (rowB + kt + rk) * AD + hOff + dk;
    const __bf16* g1p = Km + (rowB + kt + rk + 16) * AD + hOff + dk;
    *(bf16x8*)(&sK[buf][rk * LDK + dk])        = *(const bf16x8*)g0p;
    *(bf16x8*)(&sK[buf][(rk + 16) * LDK + dk]) = *(const bf16x8*)g1p;
#endif
  };

  // prologue
  loadV(0);
  stageK(0, 0);
  storeV(0);
  stage_wait();
  __syncthreads();

  int cur = 0;
  for (int kt = 0; kt < AS; kt += 32) {
    bool nxt = (kt + 32) < AS;
    if (nxt) {
      stageK(cur ^ 1, kt + 32);
      loadV(kt + 32);
    }

    // scores S[32q x 32k]
    v16bf bk[2][2];
#pragma unroll
    for (int n = 0; n < 2; ++n)
#pragma unroll
      for (int kc = 0; kc < 2; ++kc)
        bk[n][kc] = load_frag_b(sK[cur], LDK, n * 16, kc * 32);

    v8f sc[2][2];
#pragma unroll
    for (int m = 0; m < 2; ++m)
#pragma unroll
      for (int n = 0; n < 2; ++n) {
        v8f s = vzero8();
        s = wmma_bf16(qf[m][0], bk[n][0], s);
        s = wmma_bf16(qf[m][1], bk[n][1], s);
        sc[m][n] = s;
      }

    // online softmax; row r+8*half lives within one 16-lane half-wave
#pragma unroll
    for (int m = 0; m < 2; ++m) {
      float p0[8], p1[8];
#pragma unroll
      for (int r = 0; r < 8; ++r) {
        float s0 = sc[m][0][r], s1 = sc[m][1][r];
        float cm = fmaxf(s0, s1);
#pragma unroll
        for (int msk = 1; msk <= 8; msk <<= 1) cm = fmaxf(cm, __shfl_xor(cm, msk, 32));
        float mOld = rowM[m][r];
        float mNew = fmaxf(mOld, cm);
        float rs   = __expf(mOld - mNew);
        float e0   = __expf(s0 - mNew);
        float e1   = __expf(s1 - mNew);
        float ps   = e0 + e1;
#pragma unroll
        for (int msk = 1; msk <= 8; msk <<= 1) ps += __shfl_xor(ps, msk, 32);
        rowL[m][r] = rowL[m][r] * rs + ps;
        rowM[m][r] = mNew;
#pragma unroll
        for (int n = 0; n < 4; ++n) acc[m][n][r] *= rs;
        p0[r] = e0; p1[r] = e1;
      }
      // C-layout -> per-wave LDS (LDS is in-order within a wave; no barrier)
#pragma unroll
      for (int r = 0; r < 8; ++r) {
        int row = m * 16 + r + ((lane >> 4) << 3);
        sP[wave][row * LDP + (lane & 15)]      = (__bf16)p0[r];
        sP[wave][row * LDP + 16 + (lane & 15)] = (__bf16)p1[r];
      }
    }

    // context: acc += P[32x32] * V[32x64]
    v16bf vf[4];
#pragma unroll
    for (int n = 0; n < 4; ++n) vf[n] = load_frag_b(sVt[cur], LDV, n * 16, 0);
#pragma unroll
    for (int m = 0; m < 2; ++m) {
      v16bf ap = load_frag_a(sP[wave], LDP, m * 16, 0);
#pragma unroll
      for (int n = 0; n < 4; ++n)
        acc[m][n] = wmma_bf16(ap, vf[n], acc[m][n]);
    }

    if (nxt) {
      storeV(cur ^ 1);
      stage_wait();
      __syncthreads();
      cur ^= 1;
    }
  }

  int col  = lane & 15;
  int half = lane >> 4;
#pragma unroll
  for (int m = 0; m < 2; ++m)
#pragma unroll
    for (int r = 0; r < 8; ++r) {
      float inv = 1.f / rowL[m][r];
      int qg = qWave + m * 16 + r + half * 8;
#pragma unroll
      for (int n = 0; n < 4; ++n)
        Ctx[(rowB + qg) * AD + hOff + n * 16 + col] =
            (__bf16)(acc[m][n][r] * inv);
    }
}

// ---------- host-side orchestration ----------
extern "C" void kernel_launch(void* const* d_in, const int* in_sizes, int n_in,
                              void* d_out, int out_size, void* d_ws, size_t ws_size,
                              hipStream_t stream) {
  const float* x  = (const float*)d_in[0];
  const float* WQ = (const float*)d_in[1];
  const float* WK = (const float*)d_in[2];
  const float* WV = (const float*)d_in[3];
  const float* WO = (const float*)d_in[4];
  float* out = (float*)d_out;

  const int Bn = 4, S = 2048, D = 1024;
  const int M = Bn * S;
  const size_t MD = (size_t)M * D;
  const size_t DD = (size_t)D * D;

  char* ws = (char*)d_ws;
  size_t o = 0;
  __bf16* xb  = (__bf16*)(ws + o); o += MD * 2;
  __bf16* wqb = (__bf16*)(ws + o); o += DD * 2;
  __bf16* wkb = (__bf16*)(ws + o); o += DD * 2;
  __bf16* wvb = (__bf16*)(ws + o); o += DD * 2;
  __bf16* wob = (__bf16*)(ws + o); o += DD * 2;
  __bf16* qb  = (__bf16*)(ws + o); o += MD * 2;
  __bf16* kb  = (__bf16*)(ws + o); o += MD * 2;
  __bf16* vb  = (__bf16*)(ws + o); o += MD * 2;
  __bf16* cb  = (__bf16*)(ws + o); o += MD * 2;   // ~88 MB total

  {
    int n4 = (int)(MD / 4);
    cvt_f32_bf16<<<(n4 + 255) / 256, 256, 0, stream>>>(x, xb, n4);
    int w4 = (int)(DD / 4);
    cvt_f32_bf16<<<(w4 + 255) / 256, 256, 0, stream>>>(WQ, wqb, w4);
    cvt_f32_bf16<<<(w4 + 255) / 256, 256, 0, stream>>>(WK, wkb, w4);
    cvt_f32_bf16<<<(w4 + 255) / 256, 256, 0, stream>>>(WV, wvb, w4);
    cvt_f32_bf16<<<(w4 + 255) / 256, 256, 0, stream>>>(WO, wob, w4);
  }

  dim3 gg(D / BN, M / BM);
  // Q pre-scaled by 1/sqrt(D) = 1/32 so attention needs no score scaling
  gemm_bf16<<<gg, 256, 0, stream>>>(xb, wqb, qb, nullptr, 0.03125f, M, D, D);
  gemm_bf16<<<gg, 256, 0, stream>>>(xb, wkb, kb, nullptr, 1.0f, M, D, D);
  gemm_bf16<<<gg, 256, 0, stream>>>(xb, wvb, vb, nullptr, 1.0f, M, D, D);

  attn_fused<<<dim3(S / 128, Bn * 16), 128, 0, stream>>>(qb, kb, vb, cb);

  gemm_bf16<<<gg, 256, 0, stream>>>(cb, wob, nullptr, out, 1.0f, M, D, D);
}
